// VisualEncoder_59579786330611
// MI455X (gfx1250) — matrix operand order repaired and checked by reading, not compiled, not run on hardware
//
#include <hip/hip_runtime.h>
#include <hip/hip_bf16.h>
#include <math.h>

// Dims fixed by the reference: e=128, S=1024, B=32, 3 units, 3 pairs.
#define E  128
#define SSP 1024
#define BN 32
#define FF (SSP*SSP)

typedef _Float16 v16h __attribute__((ext_vector_type(16)));
typedef _Float16 v8h  __attribute__((ext_vector_type(8)));
typedef float    v8f  __attribute__((ext_vector_type(8)));
// Pointee type of the async-LDS builtins (per hipcc diagnostic: GCC-style vector)
typedef int      v4ib __attribute__((vector_size(16)));

// ---- CDNA5 async Global->LDS path (guarded; falls back to plain copies) ----
#if defined(__has_builtin)
# if __has_builtin(__builtin_amdgcn_global_load_async_to_lds_b128) && \
     __has_builtin(__builtin_amdgcn_s_wait_asynccnt)
#  define HAVE_ASYNC_LDS 1
# endif
#endif
#ifndef HAVE_ASYNC_LDS
# define HAVE_ASYNC_LDS 0
#endif

static __device__ __forceinline__ v8f wmma_f16(v16h a, v16h b, v8f c) {
  // D = A(16x32,f16) * B(32x16,f16) + C(16x16,f32)
  return __builtin_amdgcn_wmma_f32_16x16x32_f16(false, a, false, b, (short)0, c,
                                                false, false);
}

// A operand (16x32 f16): lane m = lane&15 holds row m.
// lanes 0-15: K = k0..k0+15; lanes 16-31: K = k0+16..k0+31
static __device__ __forceinline__ v16h load_a_h(const _Float16* row, int k0, int hi) {
  const int kk = k0 + hi * 16;
  v8h lo = *(const v8h*)(row + kk);
  v8h hh = *(const v8h*)(row + kk + 8);
  v16h r;
#pragma unroll
  for (int i = 0; i < 8; ++i) { r[i] = lo[i]; r[8 + i] = hh[i]; }
  return r;
}
// Same as load_a_h but converts f32 source on the fly.
static __device__ __forceinline__ v16h load_a_f32(const float* row, int k0, int hi) {
  const int kk = k0 + hi * 16;
  v16h r;
#pragma unroll
  for (int i = 0; i < 8; ++i) {
    r[i]     = (_Float16)row[kk + i];
    r[8 + i] = (_Float16)row[kk + 8 + i];
  }
  return r;
}
// B operand (32x16 f16): lane n = lane&15 holds column n; lanes 0-15 K=k0..k0+15,
// lanes 16-31 K=k0+16..k0+31.  p points at row_of_col + k0 + hi*16 (16 contiguous).
static __device__ __forceinline__ v16h load_b_h(const _Float16* p) {
  v8h lo = *(const v8h*)p;
  v8h hh = *(const v8h*)(p + 8);
  v16h r;
#pragma unroll
  for (int i = 0; i < 8; ++i) { r[i] = lo[i]; r[8 + i] = hh[i]; }
  return r;
}

// Cooperatively stage a 128x128 f16 tile (32 KB) from global into LDS.
// 256 threads x 8 x 16B segments.  Async path keeps the copy out of VGPRs
// (GLOBAL_LOAD_ASYNC_TO_LDS_B128, tracked on ASYNCcnt).
static __device__ __forceinline__ void stage_tile(const _Float16* __restrict__ src,
                                                  _Float16* dst_lds, int tid) {
#if HAVE_ASYNC_LDS
#pragma unroll
  for (int j = 0; j < 8; ++j) {
    const int seg = tid + j * 256;  // 16-byte segment index
    __builtin_amdgcn_global_load_async_to_lds_b128(
        (__attribute__((address_space(1))) v4ib*)(v4ib*)(src + (size_t)seg * 8),
        (__attribute__((address_space(3))) v4ib*)(v4ib*)(dst_lds + (size_t)seg * 8),
        0, 0);
  }
  __builtin_amdgcn_s_wait_asynccnt(0);
#else
  const uint4* s = (const uint4*)src;
  uint4*       d = (uint4*)dst_lds;
#pragma unroll
  for (int j = 0; j < 8; ++j) d[tid + j * 256] = s[tid + j * 256];
#endif
}

// -------------------------------------------------------------------------
// Kernel 1: 1x1-conv projection + bias + L2-normalize(channel) -> f16 emb.
// 12 tensors (t): t=2m / 2m+1 -> self module m (X/Y); t=6+2p / 7+2p -> pair p.
// Block 256 (8 waves); wave w computes a 16x128 row strip via 8 WMMA tiles.
// -------------------------------------------------------------------------
__global__ __launch_bounds__(256) void k_proj_norm(
    const float* __restrict__ utils,
    const float* sp_wx, const float* sp_bx, const float* sp_wy, const float* sp_by,
    const float* pw_wx, const float* pw_bx, const float* pw_wy, const float* pw_by,
    _Float16* __restrict__ emb)
{
  __shared__ _Float16 Wl[E * E];
  __shared__ float    biasl[E];
  const int t = blockIdx.z, b = blockIdx.y, rowbase = blockIdx.x * 128;
  int u; const float* W; const float* bias;
  if (t < 6) {
    const int m = t >> 1; u = m;
    W    = (t & 1) ? sp_wy + (size_t)m * E * E : sp_wx + (size_t)m * E * E;
    bias = (t & 1) ? sp_by + m * E : sp_bx + m * E;
  } else {
    const int p = (t - 6) >> 1;
    const int ii[3] = {0, 0, 1}, jj[3] = {1, 2, 2};
    u    = (t & 1) ? jj[p] : ii[p];
    W    = (t & 1) ? pw_wy + (size_t)p * E * E : pw_wx + (size_t)p * E * E;
    bias = (t & 1) ? pw_by + p * E : pw_bx + p * E;
  }
  const int tid = threadIdx.x;
  for (int i = tid; i < E * E; i += 256) Wl[i] = (_Float16)W[i];  // f32->f16, LDS stage
  if (tid < E) biasl[tid] = bias[tid];
  __syncthreads();

  const int wave = tid >> 5, lane = tid & 31, m16 = lane & 15, hi = lane >> 4;
  const float*    src = utils + ((size_t)u * BN + b) * SSP * E + (size_t)rowbase * E;
  _Float16*       dst = emb + (size_t)t * BN * SSP * E + ((size_t)b * SSP + rowbase) * E;
  const float*   arow = src + (size_t)(wave * 16 + m16) * E;

  v8f c[8] = {};
#pragma unroll
  for (int k = 0; k < 4; ++k) {
    const int k0 = k * 32;
    v16h a = load_a_f32(arow, k0, hi);
#pragma unroll
    for (int n = 0; n < 8; ++n) {
      v16h bv = load_b_h(&Wl[(n * 16 + m16) * E + k0 + hi * 16]);  // ds_load path
      c[n] = wmma_f16(a, bv, c[n]);
    }
  }
  float biasv[8];
#pragma unroll
  for (int n = 0; n < 8; ++n) biasv[n] = biasl[n * 16 + m16];
#pragma unroll
  for (int r = 0; r < 8; ++r) {
    float ss = 0.f;
#pragma unroll
    for (int n = 0; n < 8; ++n) { float v = c[n][r] + biasv[n]; ss += v * v; }
    // butterfly within each 16-lane half (C rows r / r+8 live in separate halves)
    ss += __shfl_xor(ss, 1); ss += __shfl_xor(ss, 2);
    ss += __shfl_xor(ss, 4); ss += __shfl_xor(ss, 8);
    const float inv = 1.0f / fmaxf(sqrtf(ss), 1e-12f);
    _Float16* orow = dst + (size_t)(wave * 16 + r + hi * 8) * E;
#pragma unroll
    for (int n = 0; n < 8; ++n) {
      float v = c[n][r] + biasv[n];
      orow[n * 16 + m16] = (_Float16)(v * inv);
    }
  }
}

// -------------------------------------------------------------------------
// Kernel 2: batch stats of S[b,x,y] per 128x128 (x,y) tile; fold BN into
// scale = g*rsqrt(var+eps), shift = beta - mean*scale.  Registers hold
// sum/sumsq across the b=0..31 sweep; S-tile via f16 WMMA (K=128).
// embY y-tile is staged in LDS once per b (8x less global traffic than
// letting every wave fetch its own B operands).
// -------------------------------------------------------------------------
__global__ __launch_bounds__(256) void k_stats(
    const _Float16* __restrict__ emb,
    const float* sp_g, const float* sp_b, const float* pw_g, const float* pw_b,
    float* __restrict__ scale, float* __restrict__ shift)
{
  __shared__ _Float16 ytile[128 * E];  // 32 KB
  const int mod = blockIdx.z, x0 = blockIdx.x * 128, y0 = blockIdx.y * 128;
  const float* gp  = (mod < 3) ? sp_g + (size_t)mod * FF : pw_g + (size_t)(mod - 3) * FF;
  const float* bep = (mod < 3) ? sp_b + (size_t)mod * FF : pw_b + (size_t)(mod - 3) * FF;
  const _Float16* ex0 = emb + (size_t)(2 * mod) * BN * SSP * E;
  const _Float16* ey0 = emb + (size_t)(2 * mod + 1) * BN * SSP * E;
  const int tid = threadIdx.x, wave = tid >> 5, lane = tid & 31;
  const int m16 = lane & 15, hi = lane >> 4;

  v8f sumv[8] = {}; v8f sumq[8] = {};
  for (int b = 0; b < BN; ++b) {
    __syncthreads();  // previous iteration done with ytile
    stage_tile(ey0 + ((size_t)b * SSP + y0) * E, ytile, tid);
    __syncthreads();

    const _Float16* ex = ex0 + ((size_t)b * SSP + x0 + wave * 16 + m16) * E;
    v8f c[8] = {};
#pragma unroll
    for (int k = 0; k < 4; ++k) {
      const int k0 = k * 32;
      v16h a = load_a_h(ex, k0, hi);
#pragma unroll
      for (int n = 0; n < 8; ++n) {
        v16h bv = load_b_h(&ytile[(n * 16 + m16) * E + k0 + hi * 16]);
        c[n] = wmma_f16(a, bv, c[n]);
      }
    }
#pragma unroll
    for (int n = 0; n < 8; ++n) { sumv[n] += c[n]; sumq[n] += c[n] * c[n]; }
  }
  const float invB = 1.0f / BN;
#pragma unroll
  for (int n = 0; n < 8; ++n) {
#pragma unroll
    for (int r = 0; r < 8; ++r) {
      const float mean = sumv[n][r] * invB;
      const float var  = sumq[n][r] * invB - mean * mean;
      const float istd = rsqrtf(var + 1e-5f);
      const int x = x0 + wave * 16 + r + hi * 8;
      const int y = y0 + n * 16 + m16;
      const size_t idx = (size_t)x * SSP + y;
      const float sc = gp[idx] * istd;
      scale[(size_t)mod * FF + idx] = sc;
      shift[(size_t)mod * FF + idx] = bep[idx] - mean * sc;
    }
  }
}

// -------------------------------------------------------------------------
// Kernel 3: recompute S tiles, apply Shat = S*scale + shift, reduce into
// potX[b,x] (register + butterfly) and potY[b,y] (LDS ds_add_f32 atomics,
// written as deterministic per-xt partials).  A operands hoisted over yt;
// embY y-tile staged in LDS per yt.
// -------------------------------------------------------------------------
__global__ __launch_bounds__(256) void k_potentials(
    const _Float16* __restrict__ emb,
    const float* __restrict__ scale, const float* __restrict__ shift,
    const float* sp_mxw, const float* pw_mxw, const float* pw_myw,
    float* __restrict__ potX, float* __restrict__ potYp)
{
  __shared__ _Float16 ytile[128 * E];  // 32 KB
  __shared__ float potY_l[SSP];        // 4 KB
  const int xt = blockIdx.x, b = blockIdx.y, mod = blockIdx.z;
  const int x0 = xt * 128;
  const float* mxw = (mod < 3) ? sp_mxw + mod * SSP : pw_mxw + (mod - 3) * SSP;
  const float* myw = (mod < 3) ? sp_mxw + mod * SSP : pw_myw + (mod - 3) * SSP; // dummy for self
  const float* scm = scale + (size_t)mod * FF;
  const float* shm = shift + (size_t)mod * FF;
  const int tid = threadIdx.x, wave = tid >> 5, lane = tid & 31;
  const int m16 = lane & 15, hi = lane >> 4;
  const _Float16* exrow =
      emb + (size_t)(2 * mod) * BN * SSP * E + ((size_t)b * SSP + x0 + wave * 16 + m16) * E;
  const _Float16* ey = emb + (size_t)(2 * mod + 1) * BN * SSP * E + (size_t)b * SSP * E;

  for (int i = tid; i < SSP; i += 256) potY_l[i] = 0.f;

  v16h a4[4];
#pragma unroll
  for (int k = 0; k < 4; ++k) a4[k] = load_a_h(exrow, k * 32, hi);
  float mywv[8];
#pragma unroll
  for (int r = 0; r < 8; ++r) mywv[r] = myw[x0 + wave * 16 + r + hi * 8];

  float accX[8] = {0, 0, 0, 0, 0, 0, 0, 0};

  for (int yt = 0; yt < 8; ++yt) {
    const int y0 = yt * 128;
    __syncthreads();  // previous iteration done with ytile (also covers potY_l init)
    stage_tile(ey + (size_t)y0 * E, ytile, tid);
    if (yt < 7) {  // pull next tile's BN params toward L2/WGP$ (global_prefetch_b8)
      __builtin_prefetch(scm + (size_t)(x0 + wave * 16 + m16) * SSP + y0 + 128, 0, 1);
      __builtin_prefetch(shm + (size_t)(x0 + wave * 16 + m16) * SSP + y0 + 128, 0, 1);
    }
    __syncthreads();

    v8f c[8] = {};
#pragma unroll
    for (int k = 0; k < 4; ++k) {
      const int k0 = k * 32;
#pragma unroll
      for (int n = 0; n < 8; ++n) {
        v16h bv = load_b_h(&ytile[(n * 16 + m16) * E + k0 + hi * 16]);
        c[n] = wmma_f16(a4[k], bv, c[n]);
      }
    }
    float mxwv[8];
#pragma unroll
    for (int n = 0; n < 8; ++n) mxwv[n] = mxw[y0 + n * 16 + m16];

    float tXr[8] = {0, 0, 0, 0, 0, 0, 0, 0};
#pragma unroll
    for (int n = 0; n < 8; ++n) {
      float py = 0.f;
#pragma unroll
      for (int r = 0; r < 8; ++r) {
        const int x = x0 + wave * 16 + r + hi * 8;
        const size_t idx = (size_t)x * SSP + (y0 + n * 16 + m16);
        const float shat = c[n][r] * scm[idx] + shm[idx];
        tXr[r] += mxwv[n] * shat;
        py     += mywv[r] * shat;
      }
      py += __shfl_xor(py, 16);                     // combine row halves (same column)
      if (lane < 16) atomicAdd(&potY_l[y0 + n * 16 + lane], py);  // ds_add_f32
    }
#pragma unroll
    for (int r = 0; r < 8; ++r) {
      float tt = tXr[r];
      tt += __shfl_xor(tt, 1); tt += __shfl_xor(tt, 2);
      tt += __shfl_xor(tt, 4); tt += __shfl_xor(tt, 8);
      accX[r] += tt;
    }
  }
#pragma unroll
  for (int r = 0; r < 8; ++r)
    if (m16 == 0)
      potX[((size_t)mod * BN + b) * SSP + x0 + wave * 16 + r + hi * 8] = accX[r];
  __syncthreads();
  for (int i = tid; i < SSP; i += 256)
    potYp[(((size_t)mod * 8 + xt) * BN + b) * SSP + i] = potY_l[i];
}

// -------------------------------------------------------------------------
// Kernel 4: unary potential — WMMA e->e conv, ReLU, dot with w2, +b2.
// -------------------------------------------------------------------------
__global__ __launch_bounds__(256) void k_unary(
    const float* __restrict__ utils,
    const float* un_w1, const float* un_b1, const float* un_w2, const float* un_b2,
    float* __restrict__ unp)
{
  __shared__ _Float16 Wl[E * E];
  __shared__ float    biasl[E];
  __shared__ float    w2l[E];
  const int u = blockIdx.z, b = blockIdx.y, rowbase = blockIdx.x * 128;
  const int tid = threadIdx.x;
  const float* W = un_w1 + (size_t)u * E * E;
  for (int i = tid; i < E * E; i += 256) Wl[i] = (_Float16)W[i];
  if (tid < E) { biasl[tid] = un_b1[u * E + tid]; w2l[tid] = un_w2[u * E + tid]; }
  __syncthreads();
  const int wave = tid >> 5, lane = tid & 31, m16 = lane & 15, hi = lane >> 4;
  const float* arow =
      utils + ((size_t)u * BN + b) * SSP * E + (size_t)(rowbase + wave * 16 + m16) * E;
  v8f c[8] = {};
#pragma unroll
  for (int k = 0; k < 4; ++k) {
    v16h a = load_a_f32(arow, k * 32, hi);
#pragma unroll
    for (int n = 0; n < 8; ++n) {
      v16h bv = load_b_h(&Wl[(n * 16 + m16) * E + k * 32 + hi * 16]);
      c[n] = wmma_f16(a, bv, c[n]);
    }
  }
  float biasv[8], w2v[8];
#pragma unroll
  for (int n = 0; n < 8; ++n) {
    biasv[n] = biasl[n * 16 + m16];
    w2v[n]   = w2l[n * 16 + m16];
  }
  const float b2 = un_b2[u];
#pragma unroll
  for (int r = 0; r < 8; ++r) {
    float tt = 0.f;
#pragma unroll
    for (int n = 0; n < 8; ++n) tt += w2v[n] * fmaxf(c[n][r] + biasv[n], 0.f);
    tt += __shfl_xor(tt, 1); tt += __shfl_xor(tt, 2);
    tt += __shfl_xor(tt, 4); tt += __shfl_xor(tt, 8);
    if (m16 == 0)
      unp[((size_t)u * BN + b) * SSP + rowbase + wave * 16 + r + hi * 8] = tt + b2;
  }
}

// -------------------------------------------------------------------------
// Kernel 5: combine 4 potentials, softmax over S, attention-weighted sum.
// One block per (b, u).
// -------------------------------------------------------------------------
__global__ __launch_bounds__(256) void k_final(
    const float* __restrict__ utils,
    const float* __restrict__ unp, const float* __restrict__ potX,
    const float* __restrict__ potYp,
    const float* sp_mxb, const float* pw_mxb, const float* pw_myb,
    const float* red_w, float* __restrict__ out)
{
  __shared__ float red_l[SSP];
  __shared__ float scr[256];
  const int b = blockIdx.x, u = blockIdx.y, tid = threadIdx.x;
  const float rw0 = red_w[u * 4 + 0], rw1 = red_w[u * 4 + 1];
  const float rw2 = red_w[u * 4 + 2], rw3 = red_w[u * 4 + 3];

  for (int s = tid; s < SSP; s += 256) {
    const float p0 = unp[((size_t)u * BN + b) * SSP + s];
    const float p1 = potX[((size_t)u * BN + b) * SSP + s] + sp_mxb[u];
    float p2, p3;
    auto potYs = [&](int mod) {
      float v = 0.f;
      for (int xt = 0; xt < 8; ++xt)
        v += potYp[(((size_t)mod * 8 + xt) * BN + b) * SSP + s];
      return v;
    };
    if (u == 0) {
      p2 = potX[((size_t)3 * BN + b) * SSP + s] + pw_mxb[0];
      p3 = potX[((size_t)4 * BN + b) * SSP + s] + pw_mxb[1];
    } else if (u == 1) {
      p2 = potYs(3) + pw_myb[0];
      p3 = potX[((size_t)5 * BN + b) * SSP + s] + pw_mxb[2];
    } else {
      p2 = potYs(4) + pw_myb[1];
      p3 = potYs(5) + pw_myb[2];
    }
    red_l[s] = rw0 * p0 + rw1 * p1 + rw2 * p2 + rw3 * p3;
  }
  __syncthreads();
  float m = -3.4e38f;
  for (int s = tid; s < SSP; s += 256) m = fmaxf(m, red_l[s]);
  scr[tid] = m; __syncthreads();
  for (int off = 128; off > 0; off >>= 1) {
    if (tid < off) scr[tid] = fmaxf(scr[tid], scr[tid + off]);
    __syncthreads();
  }
  m = scr[0]; __syncthreads();
  float ssum = 0.f;
  for (int s = tid; s < SSP; s += 256) {
    const float ev = __expf(red_l[s] - m);
    red_l[s] = ev; ssum += ev;
  }
  scr[tid] = ssum; __syncthreads();
  for (int off = 128; off > 0; off >>= 1) {
    if (tid < off) scr[tid] += scr[tid + off];
    __syncthreads();
  }
  const float inv = 1.0f / scr[0];
  __syncthreads();
  if (tid < E) {
    const float* up = utils + ((size_t)u * BN + b) * SSP * E + tid;
    float acc = 0.f;
    for (int s = 0; s < SSP; ++s) acc += up[(size_t)s * E] * red_l[s];
    out[((size_t)u * BN + b) * E + tid] = acc * inv;
  }
}

// -------------------------------------------------------------------------
extern "C" void kernel_launch(void* const* d_in, const int* in_sizes, int n_in,
                              void* d_out, int out_size, void* d_ws, size_t ws_size,
                              hipStream_t stream)
{
  const float* utils  = (const float*)d_in[0];
  const float* un_w1  = (const float*)d_in[1];
  const float* un_b1  = (const float*)d_in[2];
  const float* un_w2  = (const float*)d_in[3];
  const float* un_b2  = (const float*)d_in[4];
  const float* sp_wx  = (const float*)d_in[5];
  const float* sp_bx  = (const float*)d_in[6];
  const float* sp_wy  = (const float*)d_in[7];
  const float* sp_by  = (const float*)d_in[8];
  const float* sp_g   = (const float*)d_in[9];
  const float* sp_b   = (const float*)d_in[10];
  const float* sp_mxw = (const float*)d_in[11];
  const float* sp_mxb = (const float*)d_in[12];
  const float* pw_wx  = (const float*)d_in[13];
  const float* pw_bx  = (const float*)d_in[14];
  const float* pw_wy  = (const float*)d_in[15];
  const float* pw_by  = (const float*)d_in[16];
  const float* pw_g   = (const float*)d_in[17];
  const float* pw_b   = (const float*)d_in[18];
  const float* pw_mxw = (const float*)d_in[19];
  const float* pw_mxb = (const float*)d_in[20];
  const float* pw_myw = (const float*)d_in[21];
  const float* pw_myb = (const float*)d_in[22];
  const float* red_w  = (const float*)d_in[23];

  // Workspace layout (~158 MB total)
  char* ws = (char*)d_ws;
  size_t o = 0;
  _Float16* emb   = (_Float16*)(ws + o); o += (size_t)12 * BN * SSP * E * sizeof(_Float16); // 96 MB
  float*    scale = (float*)(ws + o);    o += (size_t)6 * FF * sizeof(float);               // 24 MB
  float*    shift = (float*)(ws + o);    o += (size_t)6 * FF * sizeof(float);               // 24 MB
  float*    potX  = (float*)(ws + o);    o += (size_t)6 * BN * SSP * sizeof(float);
  float*    potYp = (float*)(ws + o);    o += (size_t)6 * 8 * BN * SSP * sizeof(float);
  float*    unp   = (float*)(ws + o);    o += (size_t)3 * BN * SSP * sizeof(float);
  (void)ws_size; (void)in_sizes; (void)n_in; (void)out_size;

  k_proj_norm<<<dim3(8, BN, 12), 256, 0, stream>>>(
      utils, sp_wx, sp_bx, sp_wy, sp_by, pw_wx, pw_bx, pw_wy, pw_by, emb);
  k_unary<<<dim3(8, BN, 3), 256, 0, stream>>>(utils, un_w1, un_b1, un_w2, un_b2, unp);
  k_stats<<<dim3(8, 8, 6), 256, 0, stream>>>(emb, sp_g, sp_b, pw_g, pw_b, scale, shift);
  k_potentials<<<dim3(8, BN, 6), 256, 0, stream>>>(
      emb, scale, shift, sp_mxw, pw_mxw, pw_myw, potX, potYp);
  k_final<<<dim3(BN, 3), 256, 0, stream>>>(
      utils, unp, potX, potYp, sp_mxb, pw_mxb, pw_myb, red_w, (float*)d_out);
}
